// Sqrtm_autograd_21165598834926
// MI455X (gfx1250) — matrix operand order repaired and checked by reading, not compiled, not run on hardware
//
#include <hip/hip_runtime.h>
#include <hip/hip_bf16.h>

// Newton-Schulz matrix sqrt for batched SPD 256x256 matrices on gfx1250.
// Compute-bound (~129 GFLOP vs ~128MB HBM traffic) -> drive the matrix pipe
// with v_wmma_f32_16x16x32_bf16 (f32 accumulate). All iterates are symmetric
// (polynomials in SPD A), so WMMA B-fragments (column access) are loaded as
// contiguous rows -- no transpose/LDS staging needed. Intermediates (4x32MB
// bf16 in d_ws) live in the 192MB L2.

typedef __attribute__((ext_vector_type(16))) __bf16        v16bf;
typedef __attribute__((ext_vector_type(8)))  float         v8f;
typedef __attribute__((ext_vector_type(4)))  unsigned int  v4u;
typedef __attribute__((ext_vector_type(8)))  unsigned int  v8u;

#define NS_DIM   256
#define NS_MAT   (NS_DIM * NS_DIM)
#define NS_BATCH 256
#define NS_ITERS 5

union FragA { v16bf h; struct { v4u lo; v4u hi; } p; };
union FragB { v16bf h; v8u u; };

__device__ __forceinline__ unsigned short f32_to_bf16_rne(float f) {
    unsigned u = __builtin_bit_cast(unsigned, f);
    u += 0x7FFFu + ((u >> 16) & 1u);          // round-to-nearest-even
    return (unsigned short)(u >> 16);
}
__device__ __forceinline__ float bf16_to_f32(unsigned short s) {
    unsigned u = ((unsigned)s) << 16;
    return __builtin_bit_cast(float, u);
}

// ---------------------------------------------------------------------------
// Kernel 1: per-batch Frobenius norm, Y0 = bf16(A/||A||), Z0 = I
// one 256-thread workgroup per batch matrix
// ---------------------------------------------------------------------------
__global__ void ns_init_kernel(const float* __restrict__ A,
                               unsigned short* __restrict__ Y0,
                               unsigned short* __restrict__ Z0,
                               float* __restrict__ norms) {
    const int b = blockIdx.x;
    const int t = threadIdx.x;
    const float* Ab = A + (size_t)b * NS_MAT;

    float s = 0.0f;
    #pragma unroll 4
    for (int i = 0; i < NS_DIM; ++i) {
        float v = Ab[i * NS_DIM + t];
        s += v * v;
    }
    __shared__ float red[256];
    __shared__ float inv_norm_s;
    red[t] = s;
    __syncthreads();
    for (int off = 128; off > 0; off >>= 1) {
        if (t < off) red[t] += red[t + off];
        __syncthreads();
    }
    if (t == 0) {
        float n = sqrtf(red[0]);
        norms[b] = n;
        inv_norm_s = 1.0f / n;
    }
    __syncthreads();
    const float inv_norm = inv_norm_s;

    unsigned short* Yb = Y0 + (size_t)b * NS_MAT;
    unsigned short* Zb = Z0 + (size_t)b * NS_MAT;
    #pragma unroll 4
    for (int i = 0; i < NS_DIM; ++i) {
        const int f = i * NS_DIM + t;
        Yb[f] = f32_to_bf16_rne(Ab[f] * inv_norm);
        Zb[f] = (i == t) ? (unsigned short)0x3F80u : (unsigned short)0u;  // bf16 1.0
    }
}

// ---------------------------------------------------------------------------
// Kernel 2: batched bf16 WMMA GEMM,  C = alpha * (A @ B) + diag * I
// A, B, C are 256x256 bf16 row-major; A and B are SYMMETRIC (SPD Newton-
// Schulz iterates), which lets us read B "columns" as contiguous rows.
//
// Block = 256 threads = 8 waves (wave32). Waves tiled 4(M) x 2(N); each wave
// owns a 32x64 output slab = 2x4 WMMA 16x16 accumulators. Block tile =
// 128x128, grid = (2, 2, 256 batches).
//
// Fragment layouts used here (contiguous k, same order in A and B):
//   A 16x32 bf16: lane m(<16): K=0..15 (32B) of row m;
//                 lane m+16:   K=16..31 of row m
//   B 32x16 bf16: lane n(<16): K=0..15 (32B) of column n;
//                 lane n+16:   K=16..31 of column n   (column == row, symm.)
// ---------------------------------------------------------------------------
__global__ void __launch_bounds__(256, 2)
ns_gemm_kernel(const unsigned short* __restrict__ Ag,
               const unsigned short* __restrict__ Bg,
               unsigned short* __restrict__ Cg,
               float alpha, float diag) {
    const int b = blockIdx.z;
    const unsigned short* Ab = Ag + (size_t)b * NS_MAT;
    const unsigned short* Bb = Bg + (size_t)b * NS_MAT;
    unsigned short*       Cb = Cg + (size_t)b * NS_MAT;

    const int lane = threadIdx.x & 31;
    const int wave = threadIdx.x >> 5;
    const int l16  = lane & 15;
    const int hi   = lane >> 4;          // 0 or 1: which half-wave

    const int wm = wave & 3;             // 0..3 -> 32-row slab within block
    const int wn = wave >> 2;            // 0..1 -> 64-col slab within block
    const int m_base = blockIdx.y * 128 + wm * 32;
    const int n_base = blockIdx.x * 128 + wn * 64;

    const v8f vzero = {0.f, 0.f, 0.f, 0.f, 0.f, 0.f, 0.f, 0.f};
    v8f acc[2][4];
    #pragma unroll
    for (int mi = 0; mi < 2; ++mi)
        #pragma unroll
        for (int ni = 0; ni < 4; ++ni)
            acc[mi][ni] = vzero;

    #pragma unroll 2
    for (int k0 = 0; k0 < NS_DIM; k0 += 32) {
        // --- A fragments (2 M-tiles), contiguous 16B+16B per lane ---
        FragA af[2];
        #pragma unroll
        for (int mi = 0; mi < 2; ++mi) {
            const int row = m_base + mi * 16 + l16;
            const unsigned short* p = Ab + row * NS_DIM + k0 + hi * 16;
            af[mi].p.lo = *reinterpret_cast<const v4u*>(p);        // K+0..7
            af[mi].p.hi = *reinterpret_cast<const v4u*>(p + 8);    // K+8..15
        }
        // --- B fragments (4 N-tiles): column col == row col (symmetric) ---
        FragB bf[4];
        #pragma unroll
        for (int ni = 0; ni < 4; ++ni) {
            const int col = n_base + ni * 16 + l16;
            const unsigned short* p = Bb + col * NS_DIM + k0 + hi * 16;
            bf[ni].u = *reinterpret_cast<const v8u*>(p);           // K..K+15
            __builtin_prefetch(p + 32, 0, 1);                      // next k-block
        }
        // --- 8 WMMAs per k-step ---
        #pragma unroll
        for (int mi = 0; mi < 2; ++mi)
            #pragma unroll
            for (int ni = 0; ni < 4; ++ni)
                acc[mi][ni] = __builtin_amdgcn_wmma_f32_16x16x32_bf16(
                    /*neg_a=*/false, af[mi].h,
                    /*neg_b=*/false, bf[ni].h,
                    /*c_mod=*/(short)0, acc[mi][ni],
                    /*reuse_a=*/false, /*reuse_b=*/false);
    }

    // --- Epilogue: C = alpha*acc (+ diag on diagonal), store bf16 ---
    // C/D layout: VGPR r holds (M = r + 8*hi, N = l16) for this 16x16 tile.
    #pragma unroll
    for (int mi = 0; mi < 2; ++mi) {
        #pragma unroll
        for (int ni = 0; ni < 4; ++ni) {
            const int row0 = m_base + mi * 16 + hi * 8;
            const int col  = n_base + ni * 16 + l16;
            #pragma unroll
            for (int r = 0; r < 8; ++r) {
                float v = alpha * acc[mi][ni][r];
                const int row = row0 + r;
                if (row == col) v += diag;
                Cb[row * NS_DIM + col] = f32_to_bf16_rne(v);
            }
        }
    }
}

// ---------------------------------------------------------------------------
// Kernel 3: out = f32(Y) * sqrt(normA[b])
// ---------------------------------------------------------------------------
__global__ void ns_final_kernel(const unsigned short* __restrict__ Y,
                                const float* __restrict__ norms,
                                float* __restrict__ out) {
    const int b = blockIdx.x;
    const int t = threadIdx.x;
    const float s = sqrtf(norms[b]);
    const unsigned short* Yb = Y + (size_t)b * NS_MAT;
    float* Ob = out + (size_t)b * NS_MAT;
    #pragma unroll 4
    for (int i = 0; i < NS_DIM; ++i) {
        const int f = i * NS_DIM + t;
        Ob[f] = bf16_to_f32(Yb[f]) * s;
    }
}

// ---------------------------------------------------------------------------
extern "C" void kernel_launch(void* const* d_in, const int* in_sizes, int n_in,
                              void* d_out, int out_size, void* d_ws, size_t ws_size,
                              hipStream_t stream) {
    const float* A = (const float*)d_in[0];
    // d_in[1] is num_iter (device int scalar) -- reference uses 5; hardcode to
    // stay graph-capture-safe (no sync/memcpy to host during capture).

    unsigned char* ws = (unsigned char*)d_ws;
    const size_t BUF = (size_t)NS_BATCH * NS_MAT * sizeof(unsigned short);  // 32 MB
    unsigned short* bufs[4] = {
        (unsigned short*)(ws + 0 * BUF),
        (unsigned short*)(ws + 1 * BUF),
        (unsigned short*)(ws + 2 * BUF),
        (unsigned short*)(ws + 3 * BUF),
    };
    float* norms = (float*)(ws + 4 * BUF);

    int y = 0, z = 1, f1 = 2, f2 = 3;   // buffer roles, rotated per iteration

    ns_init_kernel<<<dim3(NS_BATCH), dim3(256), 0, stream>>>(A, bufs[y], bufs[z], norms);

    const dim3 ggrid(2, 2, NS_BATCH);
    const dim3 gblk(256);
    for (int it = 0; it < NS_ITERS; ++it) {
        // T = 1.5*I - 0.5 * (Z @ Y)       -> f1
        ns_gemm_kernel<<<ggrid, gblk, 0, stream>>>(bufs[z], bufs[y], bufs[f1], -0.5f, 1.5f);
        // Ynew = Y @ T                    -> f2
        ns_gemm_kernel<<<ggrid, gblk, 0, stream>>>(bufs[y], bufs[f1], bufs[f2], 1.0f, 0.0f);
        // Znew = T @ Z                    -> old Y slot (Y dead after Ynew)
        ns_gemm_kernel<<<ggrid, gblk, 0, stream>>>(bufs[f1], bufs[z], bufs[y], 1.0f, 0.0f);
        const int ny = f2, nz = y, nf1 = z, nf2 = f1;
        y = ny; z = nz; f1 = nf1; f2 = nf2;
    }

    ns_final_kernel<<<dim3(NS_BATCH), dim3(256), 0, stream>>>(bufs[y], norms, (float*)d_out);
}
